// GraphAttention_1022202217252
// MI455X (gfx1250) — matrix operand order repaired and checked by reading, not compiled, not run on hardware
//
#include <hip/hip_runtime.h>
#include <hip/hip_bf16.h>

#define M_NODES 50000
#define N_NODES 50000
#define N_EDGES 800000
#define CIN     256
#define HID     256
#define HEADS   8
#define HEAD_C  32

typedef __attribute__((ext_vector_type(16))) __bf16 v16bf;
typedef __attribute__((ext_vector_type(8)))  float  v8f;

// -------------------------------------------------------------------------
// Pack a weight matrix W[Nout x K] (row-major fp32, used as Y = X @ W^T) into
// bf16 B-fragments for the WMMA 32x16 bf16 B operand, k contiguous per lane half (same order as A):
//   lane L: column n = ntile*16 + (L%16), holds 16 consecutive K values
//   starting at k0 + (L/16)*16 across its 8 VGPRs (pairs per VGPR).
// Buffer layout: [ntile][kchunk][lane][16 bf16]  -> one 32B load per lane.
// -------------------------------------------------------------------------
__global__ __launch_bounds__(256)
void pack_w_kernel(const float* __restrict__ W, __bf16* __restrict__ Bp,
                   int Nout, int K) {
  int idx = blockIdx.x * blockDim.x + threadIdx.x;
  int kTiles = K >> 5;
  int total = (Nout >> 4) * kTiles * 32;
  if (idx >= total) return;
  int lane  = idx & 31;
  int kc    = (idx >> 5) % kTiles;
  int ntile = idx / (32 * kTiles);
  int n      = ntile * 16 + (lane & 15);
  int kstart = kc * 32 + (lane >> 4) * 16;
  const float* wp = W + (long)n * K + kstart;
  __bf16* dp = Bp + (long)idx * 16;
#pragma unroll
  for (int t = 0; t < 16; ++t) dp[t] = (__bf16)wp[t];
}

__global__ __launch_bounds__(256)
void zero_kernel(float* __restrict__ p, long n) {
  long i = (long)blockIdx.x * blockDim.x + threadIdx.x;
  if (i < n) p[i] = 0.0f;
}

// -------------------------------------------------------------------------
// bf16 WMMA GEMM: Y[M x Nout] = X[M x K] @ (packed W^T) + bias, fp32 out.
// block = (32,4): 4 waves; wave w computes rows m0..m0+15, cols n0..n0+63
// (4 WMMA n-tiles). K loop over K/32 chunks. M must be a multiple of 16.
// -------------------------------------------------------------------------
__global__ __launch_bounds__(128)
void gemm_bf16_kernel(const float* __restrict__ X,
                      const __bf16* __restrict__ Bp,
                      const float* __restrict__ bias,
                      float* __restrict__ Y,
                      int K, int ldY) {
  const int lane = threadIdx.x;              // 0..31
  const int wave = threadIdx.y;              // 0..3
  const int m0   = blockIdx.x * 16;
  const int n0   = blockIdx.y * 256 + wave * 64;
  const int g    = lane >> 4;                // half-wave group
  const int row  = m0 + (lane & 15);
  const int kTiles = K >> 5;

  v8f acc[4] = {};

  const long xbase = (long)row * K;
  for (int kc = 0; kc < kTiles; ++kc) {
    const int k0 = kc << 5;
    // A fragment (16-bit A 16x32 layout): lane holds row (lane&15),
    // K = {k0+16g .. k0+16g+15} (contiguous, matching packed B).
    const float* xp = X + xbase + k0 + g * 16;
    float4 f0 = *(const float4*)(xp);
    float4 f1 = *(const float4*)(xp + 4);
    float4 f2 = *(const float4*)(xp + 8);
    float4 f3 = *(const float4*)(xp + 12);
    v16bf a;
    a[0]  = (__bf16)f0.x; a[1]  = (__bf16)f0.y; a[2]  = (__bf16)f0.z; a[3]  = (__bf16)f0.w;
    a[4]  = (__bf16)f1.x; a[5]  = (__bf16)f1.y; a[6]  = (__bf16)f1.z; a[7]  = (__bf16)f1.w;
    a[8]  = (__bf16)f2.x; a[9]  = (__bf16)f2.y; a[10] = (__bf16)f2.z; a[11] = (__bf16)f2.w;
    a[12] = (__bf16)f3.x; a[13] = (__bf16)f3.y; a[14] = (__bf16)f3.z; a[15] = (__bf16)f3.w;

#pragma unroll
    for (int nt = 0; nt < 4; ++nt) {
      const int ntile = (n0 >> 4) + nt;
      const __bf16* bp = Bp + (((long)ntile * kTiles + kc) * 32 + lane) * 16;
      v16bf b = *(const v16bf*)bp;
      // D = A x B + C   (16x16x32 bf16 -> f32)
      acc[nt] = __builtin_amdgcn_wmma_f32_16x16x32_bf16(
          false, a, false, b, (short)0, acc[nt], false, false);
    }
  }

  // Epilogue: C/D layout -> VGPR r, lanes 0-15: (M=r, N=lane); lanes 16-31:
  // (M=8+r, N=lane-16). Add bias, store fp32.
#pragma unroll
  for (int nt = 0; nt < 4; ++nt) {
    const int ncol = n0 + nt * 16 + (lane & 15);
    const float bi = bias[ncol];
#pragma unroll
    for (int r = 0; r < 8; ++r) {
      const int rrow = m0 + r + 8 * g;
      Y[(long)rrow * ldY + ncol] = acc[nt][r] + bi;
    }
  }
}

// -------------------------------------------------------------------------
// RMS-norm over 32-channel head groups, one wave32 per (node, head).
// base = node*rowStride + head*32 ; in-place.
// -------------------------------------------------------------------------
__global__ __launch_bounds__(256)
void rmsnorm_kernel(float* __restrict__ x, int numGroups, int rowStride) {
  int grp = blockIdx.x * (blockDim.x >> 5) + (threadIdx.x >> 5);
  if (grp >= numGroups) return;
  int lane = threadIdx.x & 31;
  int n = grp >> 3, h = grp & 7;
  long base = (long)n * rowStride + h * 32;
  float v = x[base + lane];
  float ss = v * v;
#pragma unroll
  for (int off = 16; off; off >>= 1) ss += __shfl_xor(ss, off, 32);
  float r = rsqrtf(ss * (1.0f / 32.0f) + 1.1920929e-07f);
  x[base + lane] = v * r;
}

// -------------------------------------------------------------------------
// Edge phase 1: w[e,h] = exp(scale * <q[i,h,:], k[j,h,:]>); total[i,h] += w.
// One wave per edge: lane = h*4 + t, t covers channels t*8..t*8+7.
// -------------------------------------------------------------------------
__global__ __launch_bounds__(256)
void edge_w_kernel(const long long* __restrict__ edges,
                   const float* __restrict__ qn,     // [M,256] normalized
                   const float* __restrict__ kv,     // [N,512], k = [:,0:256] normed
                   float* __restrict__ wbuf,         // [E,8]
                   float* __restrict__ total,        // [M,8]
                   int E) {
  int e = blockIdx.x * (blockDim.x >> 5) + (threadIdx.x >> 5);
  if (e >= E) return;
  int lane = threadIdx.x & 31;
  int h = lane >> 2, t = lane & 3;
  long i = (long)edges[(long)e * 2 + 0];
  long j = (long)edges[(long)e * 2 + 1];
  const float* qp = qn + i * 256 + h * 32 + t * 8;
  const float* kp = kv + j * 512 + h * 32 + t * 8;
  float d = 0.0f;
#pragma unroll
  for (int c = 0; c < 8; ++c) d += qp[c] * kp[c];
  d += __shfl_xor(d, 1, 32);
  d += __shfl_xor(d, 2, 32);
  float w = __expf(d * 0.17677669529663687f);   // 1/sqrt(HEAD_C)
  if (t == 0) {
    wbuf[(long)e * 8 + h] = w;
    atomicAdd(&total[i * 8 + h], w);
  }
}

// -------------------------------------------------------------------------
// Edge phase 2: y[i,h,:] += (w[e,h]/total[i,h]) * v[j,h,:]
// -------------------------------------------------------------------------
__global__ __launch_bounds__(256)
void edge_y_kernel(const long long* __restrict__ edges,
                   const float* __restrict__ kv,     // v = [:,256:512]
                   const float* __restrict__ wbuf,
                   const float* __restrict__ total,
                   float* __restrict__ y,            // [M,256]
                   int E) {
  int e = blockIdx.x * (blockDim.x >> 5) + (threadIdx.x >> 5);
  if (e >= E) return;
  int lane = threadIdx.x & 31;
  int h = lane >> 2, t = lane & 3;
  long i = (long)edges[(long)e * 2 + 0];
  long j = (long)edges[(long)e * 2 + 1];
  float w = wbuf[(long)e * 8 + h] / total[i * 8 + h];
  const float* vp = kv + j * 512 + 256 + h * 32 + t * 8;
  float* yp = y + i * 256 + h * 32 + t * 8;
#pragma unroll
  for (int c = 0; c < 8; ++c) atomicAdd(&yp[c], w * vp[c]);
}

// -------------------------------------------------------------------------
extern "C" void kernel_launch(void* const* d_in, const int* in_sizes, int n_in,
                              void* d_out, int out_size, void* d_ws, size_t ws_size,
                              hipStream_t stream) {
  const float*     query = (const float*)d_in[0];
  const float*     keys  = (const float*)d_in[1];
  const long long* edges = (const long long*)d_in[2];
  const float*     Wq    = (const float*)d_in[3];
  const float*     bq    = (const float*)d_in[4];
  const float*     Wkv   = (const float*)d_in[5];
  const float*     bkv   = (const float*)d_in[6];
  const float*     Wo    = (const float*)d_in[7];
  const float*     bo    = (const float*)d_in[8];
  float* out = (float*)d_out;

  // ---- workspace carve-up (256B-aligned regions) ----
  char* ws = (char*)d_ws;
  size_t off = 0;
  auto carve = [&](size_t bytes) -> void* {
    off = (off + 255) & ~(size_t)255;
    void* p = ws + off;
    off += bytes;
    return p;
  };
  float* qn    = (float*)carve((size_t)M_NODES * 256 * 4);  // q normalized
  float* kvb   = (float*)carve((size_t)N_NODES * 512 * 4);  // k|v
  float* ybuf  = (float*)carve((size_t)M_NODES * 256 * 4);  // aggregated heads
  float* total = (float*)carve((size_t)M_NODES * 8 * 4);    // softmax denom
  float* wbuf  = (float*)carve((size_t)N_EDGES * 8 * 4);    // edge weights
  __bf16* pWq  = (__bf16*)carve((size_t)(HID / 16) * (CIN / 32) * 32 * 16 * 2);
  __bf16* pWkv = (__bf16*)carve((size_t)(2 * HID / 16) * (CIN / 32) * 32 * 16 * 2);
  __bf16* pWo  = (__bf16*)carve((size_t)(HID / 16) * (HID / 32) * 32 * 16 * 2);
  (void)ws_size; (void)in_sizes; (void)n_in; (void)out_size;

  // 1) pack weights to bf16 WMMA B-fragments
  {
    int tq = (HID / 16) * (CIN / 32) * 32;        // 4096
    int tk = (2 * HID / 16) * (CIN / 32) * 32;    // 8192
    int to = (HID / 16) * (HID / 32) * 32;        // 4096
    pack_w_kernel<<<(tq + 255) / 256, 256, 0, stream>>>(Wq,  pWq,  HID,     CIN);
    pack_w_kernel<<<(tk + 255) / 256, 256, 0, stream>>>(Wkv, pWkv, 2 * HID, CIN);
    pack_w_kernel<<<(to + 255) / 256, 256, 0, stream>>>(Wo,  pWo,  HID,     HID);
  }

  // 2) zero accumulators (workspace is poisoned, not zeroed)
  {
    long ny = (long)M_NODES * 256;
    long nt = (long)M_NODES * 8;
    zero_kernel<<<(int)((ny + 255) / 256), 256, 0, stream>>>(ybuf, ny);
    zero_kernel<<<(int)((nt + 255) / 256), 256, 0, stream>>>(total, nt);
  }

  // 3) q = query @ Wq^T + bq          [50000 x 256]
  {
    dim3 grid(M_NODES / 16, HID / 256), block(32, 4);
    gemm_bf16_kernel<<<grid, block, 0, stream>>>(query, pWq, bq, qn, CIN, HID);
  }
  // 4) kv = keys @ Wkv^T + bkv        [50000 x 512]
  {
    dim3 grid(N_NODES / 16, 2 * HID / 256), block(32, 4);
    gemm_bf16_kernel<<<grid, block, 0, stream>>>(keys, pWkv, bkv, kvb, CIN, 2 * HID);
  }

  // 5) rms_norm per head: q (all), k (first 256 of each kv row)
  {
    int gq = M_NODES * HEADS;
    rmsnorm_kernel<<<(gq + 7) / 8, 256, 0, stream>>>(qn,  gq, 256);
    int gk = N_NODES * HEADS;
    rmsnorm_kernel<<<(gk + 7) / 8, 256, 0, stream>>>(kvb, gk, 512);
  }

  // 6) edge attention logits + exp + per-destination totals
  edge_w_kernel<<<(N_EDGES + 7) / 8, 256, 0, stream>>>(edges, qn, kvb, wbuf, total, N_EDGES);

  // 7) normalized weighted scatter of v into y
  edge_y_kernel<<<(N_EDGES + 7) / 8, 256, 0, stream>>>(edges, kvb, wbuf, total, ybuf, N_EDGES);

  // 8) out = y @ Wo^T + bo            [50000 x 256]
  {
    dim3 grid(M_NODES / 16, HID / 256), block(32, 4);
    gemm_bf16_kernel<<<grid, block, 0, stream>>>(ybuf, pWo, bo, out, HID, HID);
  }
}
